// SODEDecoderBlock_51238959841570
// MI455X (gfx1250) — matrix operand order repaired and checked by reading, not compiled, not run on hardware
//
#include <hip/hip_runtime.h>
#include <math.h>

// ---------------- problem constants ----------------
#define NF_    128
#define CS_    192          // C_STATE
#define CH_    96           // C_STATE/2
#define B_     4
#define H1_    64
#define W1_    64
#define H2_    128
#define W2_    128
#define HW1_   (H1_*W1_)    // 4096
#define HW2_   (H2_*W2_)    // 16384
#define NPIX1_ (B_*HW1_)    // 16384
#define NPIX2_ (B_*HW2_)    // 65536
#define EPS_   1e-5f
#define DT_    0.25f

typedef __attribute__((ext_vector_type(16))) __bf16 v16bf;
typedef __attribute__((ext_vector_type(8)))  float  v8f;

struct __attribute__((aligned(16))) U128 { unsigned int x, y, z, w; };
union Frag { U128 q[2]; v16bf v; };

__device__ __forceinline__ unsigned short f2bf(float f) {
  unsigned int u = __float_as_uint(f);
  u += 0x7fffu + ((u >> 16) & 1u);        // round-to-nearest-even
  return (unsigned short)(u >> 16);
}
__device__ __forceinline__ float softplus_f(float x) {
  return fmaxf(x, 0.f) + log1pf(expf(-fabsf(x)));
}

// ---------------- weight prepack: OIHW f32 -> [co][tap*Cin+ci] bf16 ----------------
__global__ void prepack3x3(const float* __restrict__ src, unsigned short* __restrict__ dst,
                           int Cin, int n) {
  int i = blockIdx.x * 256 + threadIdx.x; if (i >= n) return;
  int t  = i % 9;
  int ci = (i / 9) % Cin;
  int co = i / (9 * Cin);
  dst[(size_t)co * 9 * Cin + t * Cin + ci] = f2bf(src[i]);
}
__global__ void prepack1x1(const float* __restrict__ src, unsigned short* __restrict__ dst, int n) {
  int i = blockIdx.x * 256 + threadIdx.x; if (i >= n) return;
  dst[i] = f2bf(src[i]);
}

// ---------------- instance-norm stats for NCHW x (contiguous per (b,c)) ----------------
__global__ void stats_xc(const float* __restrict__ x, float2* __restrict__ st) {
  __shared__ float s1[256], s2[256];
  const float* p = x + (size_t)blockIdx.x * HW1_;
  float a = 0.f, q = 0.f;
  for (int i = threadIdx.x; i < HW1_; i += 256) { float v = p[i]; a += v; q += v * v; }
  s1[threadIdx.x] = a; s2[threadIdx.x] = q; __syncthreads();
  for (int o = 128; o > 0; o >>= 1) {
    if ((int)threadIdx.x < o) { s1[threadIdx.x] += s1[threadIdx.x + o]; s2[threadIdx.x] += s2[threadIdx.x + o]; }
    __syncthreads();
  }
  if (threadIdx.x == 0) {
    float m = s1[0] * (1.f / HW1_);
    float v = s2[0] * (1.f / HW1_) - m * m;
    st[blockIdx.x] = make_float2(m, rsqrtf(v + EPS_));
  }
}

// normalize x -> bf16 NHWC, and copy raw x into y64 NHWC channels [0,128)
__global__ void packA(const float* __restrict__ x, const float2* __restrict__ st,
                      unsigned short* __restrict__ xn, float* __restrict__ y64, int n) {
  int i = blockIdx.x * 256 + threadIdx.x; if (i >= n) return;
  int hw = i % HW1_;
  int c  = (i / HW1_) % NF_;
  int b  = i / (HW1_ * NF_);
  float2 s = st[b * NF_ + c];
  float v = x[i];
  size_t p = (size_t)b * HW1_ + hw;
  xn[p * NF_ + c]  = f2bf((v - s.x) * s.y);
  y64[p * 256 + c] = v;
}

// ---------------- stats for NHWC activations (atomic partial reduction) ----------------
__global__ void zero2(float* a, float* b, int n) {
  int i = blockIdx.x * 256 + threadIdx.x; if (i >= n) return;
  a[i] = 0.f; b[i] = 0.f;
}
__global__ void stats_part(const float* __restrict__ y, float* __restrict__ ssum,
                           float* __restrict__ ssq, int C, int HW) {
  int c  = threadIdx.x;            // blockDim.x == C (96 or 192, multiples of 32)
  int b  = blockIdx.y;
  int p0 = blockIdx.x * 256;
  const float* base = y + ((size_t)b * HW + p0) * C + c;
  float a = 0.f, q = 0.f;
  for (int p = 0; p < 256; ++p) { float v = base[(size_t)p * C]; a += v; q += v * v; }
  atomicAdd(&ssum[b * C + c], a);
  atomicAdd(&ssq[b * C + c], q);
}
__global__ void stats_fin(const float* __restrict__ ssum, const float* __restrict__ ssq,
                          float2* __restrict__ st, int n, float inv) {
  int i = blockIdx.x * 256 + threadIdx.x; if (i >= n) return;
  float m = ssum[i] * inv;
  float v = ssq[i] * inv - m * m;
  st[i] = make_float2(m, rsqrtf(v + EPS_));
}
__global__ void normpack(const float* __restrict__ y, const float2* __restrict__ st,
                         unsigned short* __restrict__ out, int C, int HWC, int n) {
  int i = blockIdx.x * 256 + threadIdx.x; if (i >= n) return;
  int c = i % C;
  int b = i / HWC;
  float2 s = st[b * C + c];
  out[i] = f2bf((y[i] - s.x) * s.y);
}

// ---------------- implicit-GEMM conv via bf16 WMMA (f32 accumulate) ----------------
// Per-wave tile: M=32 (two A fragments) x N=64 (four 16-pixel groups) -> 8 accumulators.
// Every B fragment feeds 2 WMMAs, halving L2 traffic per FLOP vs M=16.
// in : NHWC bf16 [B][H][W][CIN]
// wp : [Cout][NTAPS*CIN] bf16, K = (kh*3+kw)*CIN + ci
// out: NHWC f32, channel stride outStride, written at coOff+co with bias+softplus
template <int H, int W, int CIN, int NTAPS>
__global__ __launch_bounds__(128) void conv_wmma(
    const unsigned short* __restrict__ in, const unsigned short* __restrict__ wp,
    const float* __restrict__ bias, float* __restrict__ out,
    int outStride, int coOff) {
  constexpr int KTOT = NTAPS * CIN;
  constexpr int HW   = H * W;
  const int lane = threadIdx.x & 31;
  const int wave = threadIdx.x >> 5;
  const int n16  = lane & 15;
  const int g    = lane >> 4;
  const int cobase = blockIdx.y << 5;      // 32 output channels per block (y)

  // 64 consecutive pixels per wave; start aligned to 64 => never crosses a row
  const int p0 = blockIdx.x * 256 + wave * 64;
  const int w0 = p0 % W;            // constexpr power-of-two -> mask
  const int hb = p0 / W;            // -> shift
  const int h  = hb % H;
  const int b  = hb / H;
  const unsigned short* inb   = in + (size_t)b * HW * CIN;
  const unsigned short* wrow0 = wp + (size_t)(cobase + n16) * KTOT;
  const unsigned short* wrow1 = wrow0 + (size_t)16 * KTOT;

  v8f acc[2][4] = {};

#pragma unroll
  for (int tap = 0; tap < NTAPS; ++tap) {
    constexpr bool IS3 = (NTAPS == 9);
    const int dh = IS3 ? (tap / 3 - 1) : 0;
    const int dw = IS3 ? (tap % 3 - 1) : 0;
    const int hh = h + dh;
    const bool hok = (unsigned)hh < (unsigned)H;
    const unsigned short* rowp  = inb + (size_t)hh * W * CIN + (g << 3);
    const unsigned short* wtap0 = wrow0 + tap * CIN + (g << 3);
    const unsigned short* wtap1 = wrow1 + tap * CIN + (g << 3);

#pragma unroll
    for (int cc = 0; cc < (CIN >> 5); ++cc) {
      // A fragments: lane (m,g) holds K = {8g..8g+7} then {16+8g..16+8g+7} (ISA 7.12.2)
      Frag af0, af1;
      {
        const unsigned short* ap0 = wtap0 + (cc << 5);
        const unsigned short* ap1 = wtap1 + (cc << 5);
        af0.q[0] = *(const U128*)(ap0);
        af0.q[1] = *(const U128*)(ap0 + 16);
        af1.q[0] = *(const U128*)(ap1);
        af1.q[1] = *(const U128*)(ap1 + 16);
      }
#pragma unroll
      for (int j = 0; j < 4; ++j) {
        // B fragment: lane (n,g) holds column n, K = {8g..8g+7} then {16+8g..16+8g+7} (matches A)
        Frag bfr;
        const int ww = w0 + j * 16 + n16 + dw;
        if (hok && (unsigned)ww < (unsigned)W) {
          const unsigned short* sp = rowp + (size_t)ww * CIN + (cc << 5);
          bfr.q[0] = *(const U128*)(sp);
          bfr.q[1] = *(const U128*)(sp + 16);
        } else {
          bfr.q[0] = U128{0u, 0u, 0u, 0u};
          bfr.q[1] = U128{0u, 0u, 0u, 0u};
        }
        acc[0][j] = __builtin_amdgcn_wmma_f32_16x16x32_bf16(
            false, af0.v, false, bfr.v, (short)0, acc[0][j], false, false);
        acc[1][j] = __builtin_amdgcn_wmma_f32_16x16x32_bf16(
            false, af1.v, false, bfr.v, (short)0, acc[1][j], false, false);
      }
    }
  }

  // D layout: lane (n,g), VGPR r -> (co = 16t + 8g + r, pixel n): 8 consecutive channels
#pragma unroll
  for (int t = 0; t < 2; ++t) {
    const int co0 = cobase + 16 * t + (g << 3);
    float bv[8];
#pragma unroll
    for (int r = 0; r < 8; ++r) bv[r] = bias[co0 + r];
#pragma unroll
    for (int j = 0; j < 4; ++j) {
      const int pix = p0 + j * 16 + n16;
      float* dst = out + (size_t)pix * outStride + coOff + co0;
#pragma unroll
      for (int r = 0; r < 8; ++r) {
        float v = acc[t][j][r] + bv[r];
        dst[r] = softplus_f(v);
      }
    }
  }
}

// ---------------- bilinear 2x upsample of y64 (256ch) + concat skip (128ch) -> bf16 NHWC 384 ----
__global__ void upsamp(const float* __restrict__ y64, const float* __restrict__ skip,
                       unsigned short* __restrict__ dst, int n) {
  int i = blockIdx.x * 256 + threadIdx.x; if (i >= n) return;
  int c = i % 384; int rest = i / 384;
  int w2 = rest % W2_; rest /= W2_;
  int h2 = rest % H2_;
  int b  = rest / H2_;
  float v;
  if (c < 256) {
    float sh = (h2 + 0.5f) * 0.5f - 0.5f;
    float sw = (w2 + 0.5f) * 0.5f - 0.5f;
    int h0 = (int)floorf(sh), w0 = (int)floorf(sw);
    float fh = sh - (float)h0, fw = sw - (float)w0;
    int h0c = min(max(h0, 0), H1_ - 1), h1c = min(max(h0 + 1, 0), H1_ - 1);
    int w0c = min(max(w0, 0), W1_ - 1), w1c = min(max(w0 + 1, 0), W1_ - 1);
    const float* base = y64 + (size_t)b * HW1_ * 256 + c;
    float v00 = base[(size_t)(h0c * W1_ + w0c) * 256];
    float v01 = base[(size_t)(h0c * W1_ + w1c) * 256];
    float v10 = base[(size_t)(h1c * W1_ + w0c) * 256];
    float v11 = base[(size_t)(h1c * W1_ + w1c) * 256];
    v = (1.f - fh) * ((1.f - fw) * v00 + fw * v01) + fh * ((1.f - fw) * v10 + fw * v11);
  } else {
    v = skip[(((size_t)b * NF_ + (c - 256)) * H2_ + h2) * W2_ + w2];
  }
  dst[i] = f2bf(v);
}

// ---------------- ODE plumbing (float4-vectorized: bandwidth-bound) ----------------
__global__ void copyv4(const float* __restrict__ y, float* __restrict__ k, int n4) {
  int i = blockIdx.x * 256 + threadIdx.x; if (i >= n4) return;
  int c4 = i % (CH_ / 4);                  // float4 index within 96-ch half
  int p  = i / (CH_ / 4);
  const float4* src = (const float4*)(y + (size_t)p * CS_ + CH_) + c4;
  float4*       dst = (float4*)(k + (size_t)p * CS_) + c4;
  *dst = *src;
}
__global__ void combine2(float4* __restrict__ d, const float4* __restrict__ a,
                         const float4* __restrict__ b, float al, int n4) {
  int i = blockIdx.x * 256 + threadIdx.x; if (i >= n4) return;
  float4 av = a[i], bv = b[i];
  d[i] = make_float4(av.x + al * bv.x, av.y + al * bv.y,
                     av.z + al * bv.z, av.w + al * bv.w);
}
__global__ void accum(float4* __restrict__ d, const float4* __restrict__ a, float al, int n4) {
  int i = blockIdx.x * 256 + threadIdx.x; if (i >= n4) return;
  float4 dv = d[i], av = a[i];
  d[i] = make_float4(dv.x + al * av.x, dv.y + al * av.y,
                     dv.z + al * av.z, dv.w + al * av.w);
}
__global__ void final3(float4* __restrict__ s, const float4* __restrict__ a,
                       const float4* __restrict__ k, float al, int n4) {
  int i = blockIdx.x * 256 + threadIdx.x; if (i >= n4) return;
  float4 sv = s[i], av = a[i], kv = k[i];
  s[i] = make_float4(sv.x + al * (av.x + kv.x), sv.y + al * (av.y + kv.y),
                     sv.z + al * (av.z + kv.z), sv.w + al * (av.w + kv.w));
}
__global__ void unpack_out(const float* __restrict__ s, float* __restrict__ out, int n) {
  int i = blockIdx.x * 256 + threadIdx.x; if (i >= n) return;
  int hw = i % HW2_;
  int c  = (i / HW2_) % CS_;
  int b  = i / (HW2_ * CS_);
  out[i] = s[((size_t)b * HW2_ + hw) * CS_ + c];
}

// ---------------- host orchestration ----------------
extern "C" void kernel_launch(void* const* d_in, const int* in_sizes, int n_in,
                              void* d_out, int out_size, void* d_ws, size_t ws_size,
                              hipStream_t stream) {
  (void)in_sizes; (void)n_in; (void)out_size; (void)ws_size;
  const float* x    = (const float*)d_in[0];
  const float* skip = (const float*)d_in[1];
  const float* iv_w = (const float*)d_in[2];
  const float* iv_b = (const float*)d_in[3];
  const float* up_w = (const float*)d_in[4];
  const float* up_b = (const float*)d_in[5];
  const float* f1_w = (const float*)d_in[6];
  const float* f1_b = (const float*)d_in[7];
  const float* f2_w = (const float*)d_in[8];
  const float* f2_b = (const float*)d_in[9];
  float* outp = (float*)d_out;

  char* wsp = (char*)d_ws;
  auto alloc = [&](size_t bytes) -> void* {
    void* p = (void*)wsp;
    wsp += (bytes + 255) & ~(size_t)255;
    return p;
  };
  unsigned short* wivp = (unsigned short*)alloc((size_t)NF_ * 9 * NF_ * 2);
  unsigned short* wupp = (unsigned short*)alloc((size_t)CS_ * 384 * 2);
  unsigned short* wf1p = (unsigned short*)alloc((size_t)CH_ * 9 * CS_ * 2);
  unsigned short* wf2p = (unsigned short*)alloc((size_t)CH_ * 9 * CH_ * 2);
  float2* stA  = (float2*)alloc((size_t)B_ * NF_ * sizeof(float2));
  float*  ssum = (float*) alloc((size_t)B_ * CS_ * 4);
  float*  ssq  = (float*) alloc((size_t)B_ * CS_ * 4);
  float2* st   = (float2*)alloc((size_t)B_ * CS_ * sizeof(float2));
  unsigned short* xn   = (unsigned short*)alloc((size_t)NPIX1_ * NF_ * 2);
  float*          y64  = (float*)         alloc((size_t)NPIX1_ * 256 * 4);
  unsigned short* upin = (unsigned short*)alloc((size_t)NPIX2_ * 384 * 2);
  float*          s    = (float*)         alloc((size_t)NPIX2_ * CS_ * 4);
  float*          ytmp = (float*)         alloc((size_t)NPIX2_ * CS_ * 4);
  float*          kb   = (float*)         alloc((size_t)NPIX2_ * CS_ * 4);
  float*          accb = (float*)         alloc((size_t)NPIX2_ * CS_ * 4);
  unsigned short* yn   = (unsigned short*)alloc((size_t)NPIX2_ * CS_ * 2);
  float*          u    = (float*)         alloc((size_t)NPIX2_ * CH_ * 4);
  unsigned short* un   = (unsigned short*)alloc((size_t)NPIX2_ * CH_ * 2);

  auto g1 = [](int n) { return dim3((unsigned)((n + 255) / 256)); };

  // weight prepack (bf16, K = tap*Cin + ci)
  prepack3x3<<<g1(NF_ * NF_ * 9), 256, 0, stream>>>(iv_w, wivp, NF_, NF_ * NF_ * 9);
  prepack1x1<<<g1(CS_ * 384),     256, 0, stream>>>(up_w, wupp, CS_ * 384);
  prepack3x3<<<g1(CH_ * CS_ * 9), 256, 0, stream>>>(f1_w, wf1p, CS_, CH_ * CS_ * 9);
  prepack3x3<<<g1(CH_ * CH_ * 9), 256, 0, stream>>>(f2_w, wf2p, CH_, CH_ * CH_ * 9);

  // ---- stage A: v0 = softplus(conv(IN(x))), y64 = concat(x, v0) (NHWC) ----
  stats_xc<<<dim3(B_ * NF_), 256, 0, stream>>>(x, stA);
  packA<<<g1(NPIX1_ * NF_), 256, 0, stream>>>(x, stA, xn, y64, NPIX1_ * NF_);
  conv_wmma<H1_, W1_, NF_, 9><<<dim3(NPIX1_ / 256, NF_ / 32), 128, 0, stream>>>(
      xn, wivp, iv_b, y64, 256, NF_);
  // ---- upsample 2x + concat skip -> 384ch bf16 NHWC; then 1x1 conv -> state s ----
  upsamp<<<g1(NPIX2_ * 384), 256, 0, stream>>>(y64, skip, upin, NPIX2_ * 384);
  conv_wmma<H2_, W2_, 384, 1><<<dim3(NPIX2_ / 256, CS_ / 32), 128, 0, stream>>>(
      upin, wupp, up_b, s, CS_, 0);

  // ---- ODE right-hand side: k = f(y) ----
  auto feval = [&](const float* yin, float* kout) {
    zero2<<<g1(B_ * CS_), 256, 0, stream>>>(ssum, ssq, B_ * CS_);
    stats_part<<<dim3(HW2_ / 256, B_), CS_, 0, stream>>>(yin, ssum, ssq, CS_, HW2_);
    stats_fin<<<g1(B_ * CS_), 256, 0, stream>>>(ssum, ssq, st, B_ * CS_, 1.f / HW2_);
    normpack<<<g1(NPIX2_ * CS_), 256, 0, stream>>>(yin, st, yn, CS_, HW2_ * CS_, NPIX2_ * CS_);
    conv_wmma<H2_, W2_, CS_, 9><<<dim3(NPIX2_ / 256, CH_ / 32), 128, 0, stream>>>(
        yn, wf1p, f1_b, u, CH_, 0);
    zero2<<<g1(B_ * CH_), 256, 0, stream>>>(ssum, ssq, B_ * CH_);
    stats_part<<<dim3(HW2_ / 256, B_), CH_, 0, stream>>>(u, ssum, ssq, CH_, HW2_);
    stats_fin<<<g1(B_ * CH_), 256, 0, stream>>>(ssum, ssq, st, B_ * CH_, 1.f / HW2_);
    normpack<<<g1(NPIX2_ * CH_), 256, 0, stream>>>(u, st, un, CH_, HW2_ * CH_, NPIX2_ * CH_);
    conv_wmma<H2_, W2_, CH_, 9><<<dim3(NPIX2_ / 256, CH_ / 32), 128, 0, stream>>>(
        un, wf2p, f2_b, kout, CS_, CH_);
    copyv4<<<g1(NPIX2_ * CH_ / 4), 256, 0, stream>>>(yin, kout, NPIX2_ * CH_ / 4);
  };

  const int NTOT  = NPIX2_ * CS_;
  const int NTOT4 = NTOT / 4;
  float4* s4 = (float4*)s; float4* y4 = (float4*)ytmp;
  float4* k4 = (float4*)kb; float4* a4 = (float4*)accb;
  for (int step = 0; step < 4; ++step) {
    feval(s, kb);                                                          // k1
    combine2<<<g1(NTOT4), 256, 0, stream>>>(a4, k4, k4, 0.f, NTOT4);       // acc = k1
    combine2<<<g1(NTOT4), 256, 0, stream>>>(y4, s4, k4, 0.5f * DT_, NTOT4);
    feval(ytmp, kb);                                                       // k2
    accum<<<g1(NTOT4), 256, 0, stream>>>(a4, k4, 2.f, NTOT4);
    combine2<<<g1(NTOT4), 256, 0, stream>>>(y4, s4, k4, 0.5f * DT_, NTOT4);
    feval(ytmp, kb);                                                       // k3
    accum<<<g1(NTOT4), 256, 0, stream>>>(a4, k4, 2.f, NTOT4);
    combine2<<<g1(NTOT4), 256, 0, stream>>>(y4, s4, k4, DT_, NTOT4);
    feval(ytmp, kb);                                                       // k4
    final3<<<g1(NTOT4), 256, 0, stream>>>(s4, a4, k4, DT_ / 6.f, NTOT4);
  }
  unpack_out<<<g1(NTOT), 256, 0, stream>>>(s, outp, NTOT);
}
